// CombinedModel_2869038154473
// MI455X (gfx1250) — matrix operand order repaired and checked by reading, not compiled, not run on hardware
//
#include <hip/hip_runtime.h>

// ============================================================================
// CombinedModel forward for MI455X (gfx1250, wave32, WMMA).
//
// All dense matmuls go through gemm_wmma_f16: 8 waves/block, each wave owns a
// 16x16 output tile via v_wmma_f32_16x16x32_f16 (f16 staged in LDS, f32
// accumulate).  Main K-loop is branch-free and double-buffered: next slab's
// global loads are issued before the WMMA on the current slab, so the
// s_wait_loadcnt lands after the matrix op.  Fragments read as ds_load_b128.
// Graph scatter/softmax use global float atomics.
//
// Workspace requirement: ~220 MB (graph-2 node buffers dominate).
// Input pointer order: setup_inputs() dict insertion order, nested dicts
// flattened in builder insertion order (99 leaves total, indices documented
// at kernel_launch).
// ============================================================================

typedef __attribute__((ext_vector_type(16))) _Float16 v16h;
typedef __attribute__((ext_vector_type(8)))  _Float16 v8h;
typedef __attribute__((ext_vector_type(2)))  _Float16 h2;
typedef __attribute__((ext_vector_type(8)))  float    v8f;

#define DEVFN __device__ __forceinline__

#define GEMM_WPB 8   // waves per block -> 16x128 output tile per block

// ---------------------------------------------------------------------------
// Generic GEMM: C[M,N] = act(A[M,K] @ B[K,N] + bias[N])
// A, B, C row-major fp32 in memory; converted to f16 in LDS for WMMA.
// grid = (ceil(M/16), ceil(N/128)); block = 256 threads (8 waves).
// act: 0 = none, 1 = relu, 2 = tanh
// ---------------------------------------------------------------------------
__global__ __launch_bounds__(256)
void gemm_wmma_f16(const float* __restrict__ A,
                   const float* __restrict__ B,
                   const float* __restrict__ bias,
                   float* __restrict__ C,
                   int M, int N, int K, int act)
{
    __shared__ _Float16 sA[2][16][32];              // [buf][m][k]
    __shared__ _Float16 sBt[2][GEMM_WPB][16][32];   // [buf][wave][n][k] (B transposed)

    const int tid  = threadIdx.x;
    const int wave = tid >> 5;
    const int lane = tid & 31;
    const int m0   = blockIdx.x * 16;
    const int n0   = (blockIdx.y * GEMM_WPB + wave) * 16;

    // A staging role: each thread owns (row, col-pair)
    const int arow = tid >> 4;          // 0..15
    const int acp  = (tid & 15) * 2;    // 0,2,..,30
    // B staging role (per wave): lane -> (n, k-group)
    const int bn  = lane & 15;          // 0..15
    const int bkg = (lane >> 4) * 16;   // 0 or 16

    const bool mFull = (m0 + 16 <= M);
    const bool nOk   = (n0 + 16 <= N);  // all model calls have N % 16 == 0
    const int  kMain = mFull ? (K & ~31) : 0;

    const int hi = lane >> 4;
    const int lr = lane & 15;

    v8f acc = {};

    // -- pipeline helpers ---------------------------------------------------
    auto fetch = [&](int kk, float& ax, float& ay, float* br) {
        const float* ap = A + (size_t)(m0 + arow) * K + kk + acp;
        ax = ap[0];
        ay = ap[1];
        if (nOk) {
            const float* bp = B + (size_t)(kk + bkg) * N + n0 + bn;
#pragma unroll
            for (int k2 = 0; k2 < 16; ++k2) br[k2] = bp[(size_t)k2 * N];
        } else {
#pragma unroll
            for (int k2 = 0; k2 < 16; ++k2) br[k2] = 0.0f;
        }
    };
    auto stage = [&](int buf, float ax, float ay, const float* br) {
        h2 apack = {(_Float16)ax, (_Float16)ay};
        *(h2*)&sA[buf][arow][acp] = apack;
        v8h b0, b1;
#pragma unroll
        for (int k2 = 0; k2 < 8; ++k2) b0[k2] = (_Float16)br[k2];
#pragma unroll
        for (int k2 = 0; k2 < 8; ++k2) b1[k2] = (_Float16)br[k2 + 8];
        *(v8h*)&sBt[buf][wave][bn][bkg]     = b0;
        *(v8h*)&sBt[buf][wave][bn][bkg + 8] = b1;
    };
    auto compute = [&](int buf) {
        // A frag: lane l: M = l&15, half j -> K = (j>>3)*16 + (l>=16?8:0) + (j&7)
        // B frag: lane l: N = l&15, half j -> K = (j>>3)*16 + (l>=16?8:0) + (j&7)
        union { v16h v; v8h h[2]; } ua, ub;
        ua.h[0] = *(const v8h*)&sA[buf][lr][hi * 8];
        ua.h[1] = *(const v8h*)&sA[buf][lr][16 + hi * 8];
        ub.h[0] = *(const v8h*)&sBt[buf][wave][lr][hi * 8];
        ub.h[1] = *(const v8h*)&sBt[buf][wave][lr][16 + hi * 8];
        acc = __builtin_amdgcn_wmma_f32_16x16x32_f16(
            /*neg_a=*/false, ua.v, /*neg_b=*/false, ub.v,
            /*c_mod=*/(short)0, acc, /*reuse_a=*/false, /*reuse_b=*/false);
    };

    // -- main loop: branch-free, double-buffered ----------------------------
    if (kMain > 0) {
        int cur = 0;
        float ax, ay, br[16];
        fetch(0, ax, ay, br);
        stage(0, ax, ay, br);
        __syncthreads();
        for (int kk = 32; kk < kMain; kk += 32) {
            float ax2, ay2, br2[16];
            fetch(kk, ax2, ay2, br2);   // global loads in flight...
            compute(cur);               // ...while WMMA runs on current slab
            stage(cur ^ 1, ax2, ay2, br2);
            __syncthreads();
            cur ^= 1;
        }
        compute(cur);
    }

    // -- tail slabs (K remainder and/or M tail): guarded path ---------------
    for (int kk = kMain; kk < K; kk += 32) {
        __syncthreads();   // all waves done reading LDS before overwrite
        {
            int m = m0 + arow;
            float vx = 0.0f, vy = 0.0f;
            if (m < M) {
                if (kk + acp     < K) vx = A[(size_t)m * K + kk + acp];
                if (kk + acp + 1 < K) vy = A[(size_t)m * K + kk + acp + 1];
            }
            h2 apack = {(_Float16)vx, (_Float16)vy};
            *(h2*)&sA[0][arow][acp] = apack;
        }
        {
            int n = n0 + bn;
            v8h b0 = {}, b1 = {};
            if (n < N) {
#pragma unroll
                for (int k2 = 0; k2 < 8; ++k2) {
                    int k = kk + bkg + k2;
                    if (k < K) b0[k2] = (_Float16)B[(size_t)k * N + n];
                }
#pragma unroll
                for (int k2 = 0; k2 < 8; ++k2) {
                    int k = kk + bkg + 8 + k2;
                    if (k < K) b1[k2] = (_Float16)B[(size_t)k * N + n];
                }
            }
            *(v8h*)&sBt[0][wave][bn][bkg]     = b0;
            *(v8h*)&sBt[0][wave][bn][bkg + 8] = b1;
        }
        __syncthreads();
        compute(0);
    }

    // -- epilogue: D layout: vgpr r -> M = m0 + r + (hi?8:0); N = n0 + lr ---
    const int n = n0 + lr;
    if (n < N) {
        float bv = bias ? bias[n] : 0.0f;
#pragma unroll
        for (int r = 0; r < 8; ++r) {
            int m = m0 + r + hi * 8;
            if (m < M) {
                float v = acc[r] + bv;
                if (act == 1)      v = fmaxf(v, 0.0f);
                else if (act == 2) v = tanhf(v);
                C[(size_t)m * N + n] = v;
            }
        }
    }
}

// ---------------------------------------------------------------------------
// gemv: y[N] = act(x[K] @ W[K,N] + bias[N])
// ---------------------------------------------------------------------------
__global__ void gemv_kernel(const float* __restrict__ x, const float* __restrict__ W,
                            const float* __restrict__ bias, float* __restrict__ y,
                            int K, int N, int act)
{
    int j = blockIdx.x * blockDim.x + threadIdx.x;
    if (j >= N) return;
    float s = bias ? bias[j] : 0.0f;
    for (int k = 0; k < K; ++k) s += x[k] * W[(size_t)k * N + j];
    if (act == 1)      s = fmaxf(s, 0.0f);
    else if (act == 2) s = tanhf(s);
    y[j] = s;
}

__global__ void colmean_kernel(const float* __restrict__ X, float* __restrict__ out,
                               int R, int C)
{
    int j = blockIdx.x * blockDim.x + threadIdx.x;
    if (j >= C) return;
    float s = 0.0f;
    for (int r = 0; r < R; ++r) s += X[(size_t)r * C + j];
    out[j] = s / (float)R;
}

__global__ void fill_kernel(float* __restrict__ p, float v, int n)
{
    int i = blockIdx.x * blockDim.x + threadIdx.x;
    if (i < n) p[i] = v;
}

__global__ void copy_kernel(float* __restrict__ d, const float* __restrict__ s, int n)
{
    int i = blockIdx.x * blockDim.x + threadIdx.x;
    if (i < n) d[i] = s[i];
}

__global__ void relu_kernel(float* __restrict__ p, int n)
{
    int i = blockIdx.x * blockDim.x + threadIdx.x;
    if (i < n) p[i] = fmaxf(p[i], 0.0f);
}

// ---------------------------------------------------------------------------
// GCN support
// ---------------------------------------------------------------------------
__global__ void pack4_kernel(const float* __restrict__ a, const float* __restrict__ b,
                             const float* __restrict__ c, const float* __restrict__ d,
                             float* __restrict__ h, int n)
{
    int i = blockIdx.x * blockDim.x + threadIdx.x;
    if (i >= n) return;
    h[(size_t)i * 4 + 0] = a[i];
    h[(size_t)i * 4 + 1] = b[i];
    h[(size_t)i * 4 + 2] = c[i];
    h[(size_t)i * 4 + 3] = d[i];
}

__global__ void deg_accum_kernel(const int* __restrict__ dst, float* __restrict__ deg, int E)
{
    int e = blockIdx.x * blockDim.x + threadIdx.x;
    if (e < E) atomicAdd(&deg[dst[e]], 1.0f);
}

__global__ void norms_kernel(const float* __restrict__ deg, float* __restrict__ dinv,
                             float* __restrict__ selfn, int n)
{
    int i = blockIdx.x * blockDim.x + threadIdx.x;
    if (i >= n) return;
    float di = rsqrtf(deg[i]);
    dinv[i]  = di;
    selfn[i] = di * di;
}

__global__ void enorm_kernel(const int* __restrict__ src, const int* __restrict__ dst,
                             const float* __restrict__ dinv, float* __restrict__ en, int E)
{
    int e = blockIdx.x * blockDim.x + threadIdx.x;
    if (e < E) en[e] = dinv[src[e]] * dinv[dst[e]];
}

// agg[i] = hw[i]*self_norm[node] + bias[ch]   (HC = 128 channels)
__global__ void gcn_agginit_kernel(const float* __restrict__ hw, const float* __restrict__ selfn,
                                   const float* __restrict__ bias, float* __restrict__ agg,
                                   int total)
{
    int i = blockIdx.x * blockDim.x + threadIdx.x;
    if (i >= total) return;
    agg[i] = hw[i] * selfn[i >> 7] + bias[i & 127];
}

// one wave per edge; 4 channels/lane; agg[dst] += hw[src]*enorm
__global__ void gcn_scatter_kernel(const float* __restrict__ hw, const int* __restrict__ src,
                                   const int* __restrict__ dst, const float* __restrict__ en,
                                   float* __restrict__ agg, int E)
{
    int lane = threadIdx.x & 31;
    int e    = blockIdx.x * (blockDim.x >> 5) + (threadIdx.x >> 5);
    if (e >= E) return;
    int s = src[e], d = dst[e];
    float w = en[e];
    const float* hs = hw  + (size_t)s * 128;
    float*       ad = agg + (size_t)d * 128;
#pragma unroll
    for (int i = 0; i < 4; ++i) {
        int c = lane * 4 + i;
        atomicAdd(&ad[c], hs[c] * w);
    }
}

// ---------------------------------------------------------------------------
// GAT (2 heads x 128 ch)
// ---------------------------------------------------------------------------
DEVFN float leaky02(float x) { return x > 0.0f ? x : 0.2f * x; }

DEVFN void atomicMaxFloat(float* addr, float v)
{
    if (v >= 0.0f) atomicMax((int*)addr, __float_as_int(v));
    else           atomicMin((unsigned int*)addr, __float_as_uint(v));
}

__global__ void gat_asd_kernel(const float* __restrict__ hwg, const float* __restrict__ asrc,
                               const float* __restrict__ adst, float* __restrict__ a_s,
                               float* __restrict__ a_d, int n)
{
    int i = blockIdx.x * blockDim.x + threadIdx.x;
    if (i >= 2 * n) return;
    int node = i >> 1, h = i & 1;
    const float* hp = hwg + (size_t)node * 256 + h * 128;
    const float* as = asrc + h * 128;
    const float* ad = adst + h * 128;
    float s1 = 0.0f, s2 = 0.0f;
    for (int c = 0; c < 128; ++c) { float v = hp[c]; s1 += v * as[c]; s2 += v * ad[c]; }
    a_s[i] = s1; a_d[i] = s2;
}

// edges e<E are real; e in [E, E+n) are self loops
__global__ void gat_emax_kernel(const int* __restrict__ src, const int* __restrict__ dst,
                                const float* __restrict__ a_s, const float* __restrict__ a_d,
                                float* __restrict__ emax, int E, int n)
{
    int i = blockIdx.x * blockDim.x + threadIdx.x;
    if (i >= (E + n) * 2) return;
    int e = i >> 1, h = i & 1;
    int s = (e < E) ? src[e] : (e - E);
    int d = (e < E) ? dst[e] : (e - E);
    float v = leaky02(a_s[s * 2 + h] + a_d[d * 2 + h]);
    atomicMaxFloat(&emax[d * 2 + h], v);
}

__global__ void gat_den_kernel(const int* __restrict__ src, const int* __restrict__ dst,
                               const float* __restrict__ a_s, const float* __restrict__ a_d,
                               const float* __restrict__ emax, float* __restrict__ den,
                               int E, int n)
{
    int i = blockIdx.x * blockDim.x + threadIdx.x;
    if (i >= (E + n) * 2) return;
    int e = i >> 1, h = i & 1;
    int s = (e < E) ? src[e] : (e - E);
    int d = (e < E) ? dst[e] : (e - E);
    float v = leaky02(a_s[s * 2 + h] + a_d[d * 2 + h]);
    atomicAdd(&den[d * 2 + h], expf(v - emax[d * 2 + h]));
}

__global__ void bias_bcast_kernel(const float* __restrict__ b, float* __restrict__ out,
                                  int n, int C)
{
    int i = blockIdx.x * blockDim.x + threadIdx.x;
    if (i < n * C) out[i] = b[i % C];
}

// one wave per (edge incl self-loops); 8 channels/lane; head = lane>>4
__global__ void gat_scatter_kernel(const float* __restrict__ hwg, const int* __restrict__ src,
                                   const int* __restrict__ dst, const float* __restrict__ a_s,
                                   const float* __restrict__ a_d, const float* __restrict__ emax,
                                   const float* __restrict__ den, float* __restrict__ out,
                                   int E, int n)
{
    int lane = threadIdx.x & 31;
    int e    = blockIdx.x * (blockDim.x >> 5) + (threadIdx.x >> 5);
    if (e >= E + n) return;
    int s = (e < E) ? src[e] : (e - E);
    int d = (e < E) ? dst[e] : (e - E);
    int h = lane >> 4;
    float v     = leaky02(a_s[s * 2 + h] + a_d[d * 2 + h]);
    float alpha = expf(v - emax[d * 2 + h]) / den[d * 2 + h];
    const float* hs = hwg + (size_t)s * 256;
    float*       od = out + (size_t)d * 256;
#pragma unroll
    for (int i = 0; i < 8; ++i) {
        int c = lane * 8 + i;
        atomicAdd(&od[c], alpha * hs[c]);
    }
}

__global__ void segsum_kernel(const float* __restrict__ X, const int* __restrict__ seg,
                              float* __restrict__ psum, int n, int C)
{
    int lane = threadIdx.x & 31;
    int node = blockIdx.x * (blockDim.x >> 5) + (threadIdx.x >> 5);
    if (node >= n) return;
    int b = seg[node];
    const float* xp = X + (size_t)node * C;
    float*       pp = psum + (size_t)b * C;
    for (int c = lane; c < C; c += 32) atomicAdd(&pp[c], xp[c]);
}

__global__ void count_kernel(const int* __restrict__ seg, float* __restrict__ cnt, int n)
{
    int i = blockIdx.x * blockDim.x + threadIdx.x;
    if (i < n) atomicAdd(&cnt[seg[i]], 1.0f);
}

__global__ void pooled_div_kernel(const float* __restrict__ psum, const float* __restrict__ cnt,
                                  float* __restrict__ pooled, int Bn, int C, int stride, int off)
{
    int i = blockIdx.x * blockDim.x + threadIdx.x;
    if (i >= Bn * C) return;
    int b = i / C, c = i % C;
    pooled[(size_t)b * stride + off + c] = psum[i] / fmaxf(cnt[b], 1.0f);
}

// ---------------------------------------------------------------------------
// Multi-head attention over n=128 tokens.  grid=(n, nheads), block=128.
// Q/K/V/out: [n, nheads*hd] row-major.
// ---------------------------------------------------------------------------
__global__ void mha_kernel(const float* __restrict__ Q, const float* __restrict__ Kt,
                           const float* __restrict__ V, float* __restrict__ out,
                           int n, int nheads, int hd)
{
    __shared__ float prob[128];
    __shared__ float red[128];
    int row = blockIdx.x, h = blockIdx.y, tid = threadIdx.x;
    int D = nheads * hd;
    float lg = -3.4e38f;
    if (tid < n) {
        const float* qp = Q  + (size_t)row * D + h * hd;
        const float* kp = Kt + (size_t)tid * D + h * hd;
        float s = 0.0f;
        for (int c = 0; c < hd; ++c) s += qp[c] * kp[c];
        lg = s * rsqrtf((float)hd);
    }
    red[tid] = lg;
    __syncthreads();
    for (int off = 64; off > 0; off >>= 1) {
        if (tid < off) red[tid] = fmaxf(red[tid], red[tid + off]);
        __syncthreads();
    }
    float mx = red[0];
    __syncthreads();
    float ex = (tid < n) ? expf(lg - mx) : 0.0f;
    prob[tid] = ex;
    red[tid]  = ex;
    __syncthreads();
    for (int off = 64; off > 0; off >>= 1) {
        if (tid < off) red[tid] += red[tid + off];
        __syncthreads();
    }
    float inv = 1.0f / red[0];
    for (int c = tid; c < hd; c += blockDim.x) {
        float s = 0.0f;
        for (int j = 0; j < n; ++j) s += prob[j] * V[(size_t)j * D + h * hd + c];
        out[(size_t)row * D + h * hd + c] = s * inv;
    }
}

// out[row] = LN(x[row] + r[row]) * g + b      grid = rows, block = 256
__global__ void add_ln_kernel(const float* __restrict__ x, const float* __restrict__ r,
                              const float* __restrict__ g, const float* __restrict__ b,
                              float* __restrict__ out, int C)
{
    __shared__ float s1[256];
    __shared__ float s2[256];
    int row = blockIdx.x, tid = threadIdx.x;
    const float* xp = x + (size_t)row * C;
    const float* rp = r + (size_t)row * C;
    float a = 0.0f, q = 0.0f;
    for (int c = tid; c < C; c += blockDim.x) {
        float v = xp[c] + rp[c];
        a += v; q += v * v;
    }
    s1[tid] = a; s2[tid] = q;
    __syncthreads();
    for (int off = 128; off > 0; off >>= 1) {
        if (tid < off) { s1[tid] += s1[tid + off]; s2[tid] += s2[tid + off]; }
        __syncthreads();
    }
    float m   = s1[0] / (float)C;
    float var = s2[0] / (float)C - m * m;
    float inv = rsqrtf(var + 1e-6f);
    for (int c = tid; c < C; c += blockDim.x) {
        float v = xp[c] + rp[c];
        out[(size_t)row * C + c] = (v - m) * inv * g[c] + b[c];
    }
}

// cross-attn2: awV[i,j] = softmax_j(Q[i].K[j]/sqrt(128)) * V[i,j]   (elementwise *V!)
__global__ void ca2_kernel(const float* __restrict__ Q, const float* __restrict__ Km,
                           const float* __restrict__ Vm, float* __restrict__ awV)
{
    __shared__ float red[128];
    int i = blockIdx.x, tid = threadIdx.x;  // block = 128
    const float* qp = Q  + (size_t)i * 128;
    const float* kp = Km + (size_t)tid * 128;
    float s = 0.0f;
    for (int c = 0; c < 128; ++c) s += qp[c] * kp[c];
    s *= rsqrtf(128.0f);
    float lg = s;
    red[tid] = s;
    __syncthreads();
    for (int off = 64; off > 0; off >>= 1) {
        if (tid < off) red[tid] = fmaxf(red[tid], red[tid + off]);
        __syncthreads();
    }
    float mx = red[0];
    __syncthreads();
    float ex = expf(lg - mx);
    red[tid] = ex;
    __syncthreads();
    for (int off = 64; off > 0; off >>= 1) {
        if (tid < off) red[tid] += red[tid + off];
        __syncthreads();
    }
    float aw = ex / red[0];
    awV[(size_t)i * 128 + tid] = aw * Vm[(size_t)i * 128 + tid];
}

// ============================================================================
// Host orchestration
// ============================================================================
extern "C" void kernel_launch(void* const* d_in, const int* in_sizes, int n_in,
                              void* d_out, int out_size, void* d_ws, size_t ws_size,
                              hipStream_t stream)
{
    (void)in_sizes; (void)out_size; (void)ws_size;
    if (n_in < 99) return;  // unexpected flattening; bail deterministically

    const int N1 = 32768, E1 = 524288;
    const int N2 = 65536, E2 = 1048576;
    const int Bn = 128, D = 1024;
    const int SEQ1 = 5000, SEQ2 = 26000;

    // ---- input leaf indices (dict insertion order, nested dicts in builder order) ----
    // 0..3  x1,z1,y_coord1,z_coord1     4 seq1
    // 5..8  x2,z2,y_coord2,z_coord2     9 seq2
    // 10..12 g1.{W0,W,b}  13..15 g2.{W0,W,b}
    // 16..19 gat.{W,asrc,adst,b}
    // 20..47 t1: inW,inb, sa(qW,qb,kW,kb,vW,vb,oW,ob), tb.sa(8), ln1g,ln1b,ln2g,ln2b,
    //            d1W,d1b,d2W,d2b, dW,db
    // 48..75 t2 (same layout)
    // 76..94 head: ca1Wq,ca1Wk,ca1Wv, ca2Wq,ca2Wk,ca2Wv, caWq,caWk,caWv,
    //              dW,db, o1W,o1b, oW,ob, l1W,l1b, l2W,l2b
    // 95 edge_index1  96 edge_index2  97 batch1  98 batch2
    auto F = [&](int i) { return (const float*)d_in[i]; };
    auto I = [&](int i) { return (const int*)d_in[i]; };

    // ---- workspace arena ----
    float* ws = (float*)d_ws;
    size_t arena = 0;
    auto alloc = [&](size_t nf) { float* p = ws + arena; arena += (nf + 63) & ~(size_t)63; return p; };

    // persistent small buffers
    float* psum1   = alloc(128 * 256);
    float* psum2   = alloc(128 * 256);
    float* cnt1    = alloc(128);
    float* cnt2    = alloc(128);
    float* pooled  = alloc(128 * 512);
    float* sa1     = alloc(128 * 1024);
    float* sa2     = alloc(128 * 1024);
    float* scores  = alloc(2304);
    float* ca1v    = alloc(1152);
    float* den2v   = alloc(1024);
    float* ca2m    = alloc(1152);
    float* cavec   = alloc(1280);
    float* xmv     = alloc(512);
    float* x2v     = alloc(2816);
    float* scores2 = alloc(2816);
    float* t128a   = alloc(128);
    float* t128b   = alloc(128);
    float* tmean   = alloc(1024);
    float* Q2      = alloc(128 * 128);
    float* K2      = alloc(128 * 128);
    float* V2m     = alloc(128 * 128);
    float* awV     = alloc(128 * 128);

    // graph arena (sized for graph 2, reused for graph 1; towers overlay gh)
    float* gh     = alloc((size_t)N2 * 128);
    float* ghw    = alloc((size_t)N2 * 128);
    float* ghwg   = alloc((size_t)N2 * 256);
    float* gout   = alloc((size_t)N2 * 256);
    float* gdeg   = alloc(N2);
    float* gdinv  = alloc(N2);
    float* gself  = alloc(N2);
    float* genorm = alloc(E2);
    float* ga_s   = alloc((size_t)N2 * 2);
    float* ga_d   = alloc((size_t)N2 * 2);
    float* gemax  = alloc((size_t)N2 * 2);
    float* gden   = alloc((size_t)N2 * 2);

    // tower temporaries overlay the (by-then finished) graph node buffer gh
    float* Tinp = gh + 0 * 131072;
    float* TQ   = gh + 1 * 131072;
    float* TK   = gh + 2 * 131072;
    float* TV   = gh + 3 * 131072;
    float* Tatt = gh + 4 * 131072;
    float* Taux = gh + 5 * 131072;
    float* To1  = gh + 6 * 131072;
    float* Tff  = gh + 7 * 131072;
    float* To2  = gh + 8 * 131072;

    // ---- launch helpers ----
    auto gemm = [&](const float* A, const float* Bm, const float* bias, float* C,
                    int M, int Nn, int K, int act) {
        dim3 grid((M + 15) / 16, (Nn + 16 * GEMM_WPB - 1) / (16 * GEMM_WPB));
        gemm_wmma_f16<<<grid, dim3(256), 0, stream>>>(A, Bm, bias, C, M, Nn, K, act);
    };
    auto gemv = [&](const float* x, const float* W, const float* bias, float* y,
                    int K, int Nn, int act) {
        gemv_kernel<<<(Nn + 255) / 256, 256, 0, stream>>>(x, W, bias, y, K, Nn, act);
    };
    auto fillv = [&](float* p, float v, int n) {
        fill_kernel<<<(n + 255) / 256, 256, 0, stream>>>(p, v, n);
    };
    auto copyv = [&](float* dst, const float* src, int n) {
        copy_kernel<<<(n + 255) / 256, 256, 0, stream>>>(dst, src, n);
    };
    auto cmean = [&](const float* X, float* o, int R, int C) {
        colmean_kernel<<<(C + 255) / 256, 256, 0, stream>>>(X, o, R, C);
    };

    // ---------------- graphs: 16-layer GCN + GAT + mean-pool ----------------
    auto run_graph = [&](int g) {
        const int N = g == 0 ? N1 : N2;
        const int E = g == 0 ? E1 : E2;
        const int* ei    = I(g == 0 ? 95 : 96);
        const int* src   = ei;
        const int* dst   = ei + E;
        const int* batch = I(g == 0 ? 97 : 98);
        const float* W0  = F(g == 0 ? 10 : 13);
        const float* Wl  = F(g == 0 ? 11 : 14);
        const float* bl  = F(g == 0 ? 12 : 15);

        // symmetric-norm coefficients
        fillv(gdeg, 1.0f, N);
        deg_accum_kernel<<<(E + 255) / 256, 256, 0, stream>>>(dst, gdeg, E);
        norms_kernel<<<(N + 255) / 256, 256, 0, stream>>>(gdeg, gdinv, gself, N);
        enorm_kernel<<<(E + 255) / 256, 256, 0, stream>>>(src, dst, gdinv, genorm, E);

        // layer 0 (K=4)
        pack4_kernel<<<(N + 255) / 256, 256, 0, stream>>>(
            F(g == 0 ? 0 : 5), F(g == 0 ? 1 : 6), F(g == 0 ? 2 : 7), F(g == 0 ? 3 : 8), gh, N);
        gemm(gh, W0, nullptr, ghw, N, 128, 4, 0);
        gcn_agginit_kernel<<<(N * 128 + 255) / 256, 256, 0, stream>>>(ghw, gself, bl, gh, N * 128);
        gcn_scatter_kernel<<<(E + 7) / 8, 256, 0, stream>>>(ghw, src, dst, genorm, gh, E);
        relu_kernel<<<(N * 128 + 255) / 256, 256, 0, stream>>>(gh, N * 128);

        for (int l = 1; l < 16; ++l) {
            gemm(gh, Wl + (size_t)(l - 1) * 128 * 128, nullptr, ghw, N, 128, 128, 0);
            gcn_agginit_kernel<<<(N * 128 + 255) / 256, 256, 0, stream>>>(
                ghw, gself, bl + (size_t)l * 128, gh, N * 128);
            gcn_scatter_kernel<<<(E + 7) / 8, 256, 0, stream>>>(ghw, src, dst, genorm, gh, E);
            relu_kernel<<<(N * 128 + 255) / 256, 256, 0, stream>>>(gh, N * 128);
        }

        // GAT (2 heads, shared params) + mean-pool into pooled[:, g*256 : g*256+256]
        gemm(gh, F(16), nullptr, ghwg, N, 256, 128, 0);
        gat_asd_kernel<<<(2 * N + 255) / 256, 256, 0, stream>>>(ghwg, F(17), F(18), ga_s, ga_d, N);
        fillv(gemax, -3.4e38f, 2 * N);
        const int tot2 = (E + N) * 2;
        gat_emax_kernel<<<(tot2 + 255) / 256, 256, 0, stream>>>(src, dst, ga_s, ga_d, gemax, E, N);
        fillv(gden, 0.0f, 2 * N);
        gat_den_kernel<<<(tot2 + 255) / 256, 256, 0, stream>>>(src, dst, ga_s, ga_d, gemax, gden, E, N);
        bias_bcast_kernel<<<(N * 256 + 255) / 256, 256, 0, stream>>>(F(19), gout, N, 256);
        gat_scatter_kernel<<<(E + N + 7) / 8, 256, 0, stream>>>(
            ghwg, src, dst, ga_s, ga_d, gemax, gden, gout, E, N);

        float* psum = g == 0 ? psum1 : psum2;
        float* cnt  = g == 0 ? cnt1 : cnt2;
        fillv(psum, 0.0f, 128 * 256);
        fillv(cnt, 0.0f, 128);
        segsum_kernel<<<(N + 7) / 8, 256, 0, stream>>>(gout, batch, psum, N, 256);
        count_kernel<<<(N + 255) / 256, 256, 0, stream>>>(batch, cnt, N);
        pooled_div_kernel<<<(128 * 256 + 255) / 256, 256, 0, stream>>>(
            psum, cnt, pooled, 128, 256, 512, g == 0 ? 0 : 256);
    };
    run_graph(0);
    run_graph(1);

    // ---------------- sequence towers ----------------
    // denDst receives tr.mean(0) @ dW + db  (length 1024)
    auto run_tower = [&](int base, const float* seq, int SEQ, float* saOut, float* denDst) {
        gemm(seq, F(base + 0), F(base + 1), Tinp, Bn, D, SEQ, 0);
        // tower self-attention, 16 heads x 64
        gemm(Tinp, F(base + 2), F(base + 3), TQ, Bn, D, D, 0);
        gemm(Tinp, F(base + 4), F(base + 5), TK, Bn, D, D, 0);
        gemm(Tinp, F(base + 6), F(base + 7), TV, Bn, D, D, 0);
        mha_kernel<<<dim3(Bn, 16), 128, 0, stream>>>(TQ, TK, TV, Tatt, Bn, 16, 64);
        gemm(Tatt, F(base + 8), F(base + 9), saOut, Bn, D, D, 0);
        // transformer block: SA 4 heads x 256
        gemm(saOut, F(base + 10), F(base + 11), TQ, Bn, D, D, 0);
        gemm(saOut, F(base + 12), F(base + 13), TK, Bn, D, D, 0);
        gemm(saOut, F(base + 14), F(base + 15), TV, Bn, D, D, 0);
        mha_kernel<<<dim3(Bn, 4), 128, 0, stream>>>(TQ, TK, TV, Tatt, Bn, 4, 256);
        gemm(Tatt, F(base + 16), F(base + 17), Taux, Bn, D, D, 0);
        add_ln_kernel<<<Bn, 256, 0, stream>>>(saOut, Taux, F(base + 18), F(base + 19), To1, D);
        gemm(To1, F(base + 22), F(base + 23), Tff, Bn, D, D, 0);
        add_ln_kernel<<<Bn, 256, 0, stream>>>(To1, Tff, F(base + 20), F(base + 21), To2, D);
        gemm(To2, F(base + 24), F(base + 25), Taux, Bn, D, D, 0);   // tr
        cmean(Taux, tmean, Bn, D);
        gemv(tmean, F(base + 26), F(base + 27), denDst, D, D, 0);
    };
    run_tower(20, F(4), SEQ1, sa1, ca1v);       // den1 lands in ca1[0:1024]
    run_tower(48, F(9), SEQ2, sa2, den2v);

    cmean(sa1, scores + 0,    Bn, D);           // sa1.mean -> scores[0:1024]
    cmean(sa2, scores + 1024, Bn, D);           // sa2.mean -> scores[1024:2048]

    // ---------------- cross-attention head ----------------
    const int H = 76;
    // ca1 = concat(den1, den2 @ ca1Wv)
    gemv(den2v, F(H + 2), nullptr, ca1v + 1024, 1024, 128, 0);
    // ca2 = concat(sa1, softmax(QK^T)*V); only its column-mean is used
    gemm(sa1, F(H + 3), nullptr, Q2,  Bn, 128, D, 0);
    gemm(sa2, F(H + 4), nullptr, K2,  Bn, 128, D, 0);
    gemm(sa2, F(H + 5), nullptr, V2m, Bn, 128, D, 0);
    ca2_kernel<<<Bn, 128, 0, stream>>>(Q2, K2, V2m, awV);
    copyv(ca2m, scores, 1024);                  // sa1.mean
    cmean(awV, ca2m + 1024, Bn, 128);
    // ca = concat(ca1, ca2mean @ caWv)
    copyv(cavec, ca1v, 1152);
    gemv(ca2m, F(H + 8), nullptr, cavec + 1152, 1152, 128, 0);
    // cr = tanh(ca @ dW + db) -> scores[2048:2304]
    gemv(cavec, F(H + 9), F(H + 10), scores + 2048, 1280, 256, 2);
    // xm = pooled.mean(0)
    cmean(pooled, xmv, 128, 512);
    // x_2 = [xm, scores]; scores2 = [scores, xm]
    copyv(x2v, xmv, 512);
    copyv(x2v + 512, scores, 2304);
    copyv(scores2, scores, 2304);
    copyv(scores2 + 2304, xmv, 512);
    // out2 = tanh(scores2 @ o1W + o1b) @ oW + ob
    gemv(scores2, F(H + 11), F(H + 12), t128a, 2816, 128, 2);
    gemv(t128a, F(H + 13), F(H + 14), (float*)d_out + 1, 128, 1, 0);
    // out1 = relu(x_2 @ l1W + l1b) @ l2W + l2b
    gemv(x2v, F(H + 15), F(H + 16), t128b, 2816, 128, 1);
    gemv(t128b, F(H + 17), F(H + 18), (float*)d_out + 0, 128, 1, 0);
}
